// CHGNet_19713899889327
// MI455X (gfx1250) — matrix operand order repaired and checked by reading, not compiled, not run on hardware
//
#include <hip/hip_runtime.h>
#include <hip/hip_bf16.h>

#define N_ATOMS   20000
#define N_UBONDS  200000
#define N_DBONDS  400000
#define N_ANGLES  400000
#define N_GRAPHS  200
#define FEA       64
#define ATOM_CUT  5.0f
#define BOND_CUT  3.0f
#define MSG_WAVES 4

typedef __attribute__((ext_vector_type(16))) _Float16 v16h;
typedef __attribute__((ext_vector_type(8)))  _Float16 v8h;
typedef __attribute__((ext_vector_type(4)))  _Float16 v4h;
typedef __attribute__((ext_vector_type(8)))  float    v8f;

__device__ __forceinline__ v8f vzero8() {
  v8f z = {0.f,0.f,0.f,0.f,0.f,0.f,0.f,0.f};
  return z;
}

// A-fragment (16xK f16, row-major in LDS) for v_wmma_f32_16x16x32_f16.
// lanes 0-15: row M=lane, K=[kb..kb+16) contiguous
// lanes 16-31: row M=lane-16, K=[kb+16..kb+32) contiguous
__device__ __forceinline__ v16h lds_afrag(const _Float16* X, int rowStride,
                                          int kbase, int lane) {
  int r  = lane & 15;
  int hi = (lane >> 4) & 1;
  const _Float16* p = X + r * rowStride + kbase + hi * 16;
  union { v16h v; v8h h[2]; } u;
  u.h[0] = *(const v8h*)p;
  u.h[1] = *(const v8h*)(p + 8);
  return u.v;
}

// B-fragment from f16 weights stored transposed: WT[out][in] (row = out col).
// lane holds column N = nt*16 + (lane&15); K halves contiguous per lane half.
__device__ __forceinline__ v16h glb_bfrag(const void* WTv, int din, int nt,
                                          int kbase, int lane) {
  const _Float16* WT = (const _Float16*)WTv;
  int col = nt * 16 + (lane & 15);
  int hi  = (lane >> 4) & 1;
  return *(const v16h*)(WT + (size_t)col * din + kbase + hi * 16);
}

__device__ __forceinline__ v8f wmma32(v16h a, v16h b, v8f c) {
  return __builtin_amdgcn_wmma_f32_16x16x32_f16(false, a, false, b,
                                                (short)0, c, false, false);
}

// Two-layer gMLP branch: out = (X @ W0 + b0).relu() @ W1 + b1  (16 rows x 64)
// sH is per-wave scratch for re-fragmenting the hidden activation.
template<int DIN>
__device__ __forceinline__ void gmlp_branch(const _Float16* sXw, _Float16* sHw,
                                            const void* W0, const float* b0,
                                            const void* W1, const float* b1,
                                            int lane, v8f out[4]) {
  v8f acc[4];
  #pragma unroll
  for (int nt = 0; nt < 4; ++nt) acc[nt] = vzero8();
  #pragma unroll
  for (int kc = 0; kc < DIN / 32; ++kc) {
    v16h a = lds_afrag(sXw, DIN, kc * 32, lane);
    #pragma unroll
    for (int nt = 0; nt < 4; ++nt)
      acc[nt] = wmma32(a, glb_bfrag(W0, DIN, nt, kc * 32, lane), acc[nt]);
  }
  int m0 = (lane >> 4) * 8;
  int fl = lane & 15;
  #pragma unroll
  for (int nt = 0; nt < 4; ++nt) {
    float bias = b0[nt * 16 + fl];
    #pragma unroll
    for (int v = 0; v < 8; ++v) {
      float x = acc[nt][v] + bias;
      x = x > 0.f ? x : 0.f;
      sHw[(m0 + v) * FEA + nt * 16 + fl] = (_Float16)x;
    }
  }
  __syncthreads();
  v8f acc2[4];
  #pragma unroll
  for (int nt = 0; nt < 4; ++nt) acc2[nt] = vzero8();
  #pragma unroll
  for (int kc = 0; kc < 2; ++kc) {
    v16h a = lds_afrag(sHw, FEA, kc * 32, lane);
    #pragma unroll
    for (int nt = 0; nt < 4; ++nt)
      acc2[nt] = wmma32(a, glb_bfrag(W1, FEA, nt, kc * 32, lane), acc2[nt]);
  }
  #pragma unroll
  for (int nt = 0; nt < 4; ++nt) {
    float bias = b1[nt * 16 + fl];
    #pragma unroll
    for (int v = 0; v < 8; ++v) out[nt][v] = acc2[nt][v] + bias;
  }
  __syncthreads();   // protect sH before it is reused by the next branch
}

__device__ __forceinline__ void cvt_store4(_Float16* dst, int f, float4 v) {
  v4h h = { (_Float16)v.x, (_Float16)v.y, (_Float16)v.z, (_Float16)v.w };
  *(v4h*)(dst + f) = h;
}

// ---------------------------------------------------------------- weight cvt
#define MAXW 48
struct WCvtTable {
  const float* src[MAXW];
  unsigned int dstOff[MAXW];
  int din[MAXW];
  int dout[MAXW];
  int n;
};

__global__ void wcvt_kernel(WCvtTable t, void* dstBase) {
  int e = blockIdx.x;
  if (e >= t.n) return;
  const float* s = t.src[e];
  _Float16* d = (_Float16*)((char*)dstBase + t.dstOff[e]);
  int din = t.din[e], dout = t.dout[e];
  int n = din * dout;
  for (int i = threadIdx.x; i < n; i += blockDim.x) {
    int o = i / din, k = i - o * din;        // dst[o][k] = src[k][o]
    d[i] = (_Float16)s[k * dout + o];
  }
}

// ------------------------------------------------------------------- basis
__global__ void atom_embed_kernel(const int* __restrict__ z,
                                  const float* __restrict__ emb,
                                  float* __restrict__ atom_feas) {
  int i = blockIdx.x * blockDim.x + threadIdx.x;
  if (i >= N_ATOMS * FEA) return;
  int a = i >> 6, f = i & 63;
  atom_feas[i] = emb[z[a] * FEA + f];
}

__device__ __forceinline__ float cutoff_env(float d, float cutoff) {
  float r = d / cutoff;
  float r2 = r * r, r3 = r2 * r;
  float r6 = r3 * r3, r7 = r6 * r, r8 = r7 * r;
  float env = 1.f - 28.f * r6 + 48.f * r7 - 21.f * r8;  // p = 6
  return d <= cutoff ? env : 0.f;
}

__global__ void bond_basis_kernel(const float* __restrict__ dist,
                                  const float* __restrict__ fag,
                                  const float* __restrict__ fbg,
                                  const float* __restrict__ Wemb,
                                  const float* __restrict__ Wbwag,
                                  const float* __restrict__ Wbwbg,
                                  float* __restrict__ bond_feas,
                                  float* __restrict__ bw_ag,
                                  float* __restrict__ bw_bg) {
  int i = blockIdx.x * blockDim.x + threadIdx.x;
  if (i >= N_UBONDS * FEA) return;
  int b = i >> 6, f = i & 63;
  float d = dist[b];
  float ea = cutoff_env(d, ATOM_CUT) * sqrtf(2.f / ATOM_CUT) / d;
  float eb = cutoff_env(d, BOND_CUT) * sqrtf(2.f / BOND_CUT) / d;
  float bf = 0.f, wa = 0.f, wb = 0.f;
  #pragma unroll
  for (int k = 0; k < 9; ++k) {
    float ra = sinf(fag[k] * d / ATOM_CUT) * ea;
    float rb = sinf(fbg[k] * d / BOND_CUT) * eb;
    bf += ra * Wemb[k * FEA + f];
    wa += ra * Wbwag[k * FEA + f];
    wb += rb * Wbwbg[k * FEA + f];
  }
  bond_feas[i] = bf; bw_ag[i] = wa; bw_bg[i] = wb;
}

__global__ void angle_basis_kernel(const float* __restrict__ angles,
                                   const float* __restrict__ ffreq,
                                   const float* __restrict__ Wang,
                                   float* __restrict__ angle_feas) {
  int i = blockIdx.x * blockDim.x + threadIdx.x;
  if (i >= N_ANGLES * FEA) return;
  int a = i >> 6, f = i & 63;
  float t = angles[a];
  const float isp = 0.5641895835477563f;   // 1/sqrt(pi)
  float acc = 0.7071067811865475f * isp * Wang[f];
  #pragma unroll
  for (int k = 0; k < 4; ++k) {
    float s = sinf(ffreq[k] * t), c = cosf(ffreq[k] * t);
    acc += s * isp * Wang[(1 + k) * FEA + f];
    acc += c * isp * Wang[(5 + k) * FEA + f];
  }
  angle_feas[i] = acc;
}

// ---------------------------------------------------------------- atom conv
__global__ __launch_bounds__(128) void atom_msg_kernel(
    const float* __restrict__ atom_feas, const float* __restrict__ bond_feas,
    const float* __restrict__ bw_ag, const int* __restrict__ atom_graph,
    const int* __restrict__ d2u,
    const void* WTc0, const float* bc0, const void* WTc1, const float* bc1,
    const void* WTg0, const float* gb0, const void* WTg1, const float* gb1,
    float* __restrict__ new_atom) {
  __shared__ __align__(16) _Float16 sX[MSG_WAVES][16 * 192];
  __shared__ __align__(16) _Float16 sH[MSG_WAVES][16 * FEA];
  __shared__ __align__(16) float    sBW[MSG_WAVES][16 * FEA];
  __shared__ int sCtr[MSG_WAVES][16];
  const int wave = threadIdx.x >> 5;
  const int lane = threadIdx.x & 31;
  const int row0 = (blockIdx.x * MSG_WAVES + wave) * 16;
  { // stage: each lane fills half of one row (96 feats + 32 bw)
    int r = lane >> 1, h = lane & 1;
    int row = row0 + r;
    int rc = row < N_DBONDS ? row : 0;
    float scale = row < N_DBONDS ? 1.f : 0.f;
    int ctr = atom_graph[2 * rc + 0];
    int nbr = atom_graph[2 * rc + 1];
    int ub  = d2u[rc];
    if (h == 0) sCtr[wave][r] = ctr;
    const float* s0 = atom_feas + (size_t)ctr * FEA;
    const float* s1 = bond_feas + (size_t)ub  * FEA;
    const float* s2 = atom_feas + (size_t)nbr * FEA;
    _Float16* dst = &sX[wave][r * 192];
    #pragma unroll
    for (int j = 0; j < 24; ++j) {
      int f = h * 96 + j * 4;
      const float* sp = (f < 64) ? (s0 + f) : ((f < 128) ? (s1 + f - 64)
                                                         : (s2 + f - 128));
      cvt_store4(dst, f, *(const float4*)sp);
    }
    const float* bwp = bw_ag + (size_t)ub * FEA;
    #pragma unroll
    for (int j = 0; j < 8; ++j) {
      int f = h * 32 + j * 4;
      float4 v = *(const float4*)(bwp + f);
      sBW[wave][r * FEA + f + 0] = v.x * scale;
      sBW[wave][r * FEA + f + 1] = v.y * scale;
      sBW[wave][r * FEA + f + 2] = v.z * scale;
      sBW[wave][r * FEA + f + 3] = v.w * scale;
    }
  }
  __syncthreads();
  v8f core[4], gate[4];
  gmlp_branch<192>(&sX[wave][0], &sH[wave][0], WTc0, bc0, WTc1, bc1, lane, core);
  gmlp_branch<192>(&sX[wave][0], &sH[wave][0], WTg0, gb0, WTg1, gb1, lane, gate);
  int m0 = (lane >> 4) * 8, fl = lane & 15;
  #pragma unroll
  for (int nt = 0; nt < 4; ++nt) {
    #pragma unroll
    for (int v = 0; v < 8; ++v) {
      int M = m0 + v, f = nt * 16 + fl;
      float c = core[nt][v]; c = c > 0.f ? c : 0.f;
      float s = 1.f / (1.f + expf(-gate[nt][v]));
      float m = c * s * sBW[wave][M * FEA + f];
      atomicAdd(&new_atom[(size_t)sCtr[wave][M] * FEA + f], m);
    }
  }
}

// ---------------------------------------------------------------- bond conv
__global__ __launch_bounds__(128) void bond_msg_kernel(
    const float* __restrict__ bond_feas, const float* __restrict__ angle_feas,
    const float* __restrict__ atom_feas, const float* __restrict__ bw_bg,
    const int* __restrict__ bg_i, const int* __restrict__ bg_j,
    const int* __restrict__ bg_c,
    const void* WTc0, const float* bc0, const void* WTc1, const float* bc1,
    const void* WTg0, const float* gb0, const void* WTg1, const float* gb1,
    float* __restrict__ new_bond) {
  __shared__ __align__(16) _Float16 sX[MSG_WAVES][16 * 256];
  __shared__ __align__(16) _Float16 sH[MSG_WAVES][16 * FEA];
  __shared__ __align__(16) float    sBW[MSG_WAVES][16 * FEA];
  __shared__ int sCtr[MSG_WAVES][16];
  const int wave = threadIdx.x >> 5;
  const int lane = threadIdx.x & 31;
  const int row0 = (blockIdx.x * MSG_WAVES + wave) * 16;
  {
    int r = lane >> 1, h = lane & 1;
    int row = row0 + r;
    int rc = row < N_ANGLES ? row : 0;
    float scale = row < N_ANGLES ? 1.f : 0.f;
    int bi = bg_i[rc], bj = bg_j[rc], ct = bg_c[rc];
    if (h == 0) sCtr[wave][r] = bi;
    const float* s0 = bond_feas  + (size_t)bi * FEA;
    const float* s1 = bond_feas  + (size_t)bj * FEA;
    const float* s2 = angle_feas + (size_t)rc * FEA;
    const float* s3 = atom_feas  + (size_t)ct * FEA;
    _Float16* dst = &sX[wave][r * 256];
    #pragma unroll
    for (int j = 0; j < 32; ++j) {
      int f = h * 128 + j * 4;
      const float* sp = (f < 64) ? (s0 + f)
                      : (f < 128) ? (s1 + f - 64)
                      : (f < 192) ? (s2 + f - 128) : (s3 + f - 192);
      cvt_store4(dst, f, *(const float4*)sp);
    }
    const float* wa = bw_bg + (size_t)bi * FEA;
    const float* wb = bw_bg + (size_t)bj * FEA;
    #pragma unroll
    for (int j = 0; j < 8; ++j) {
      int f = h * 32 + j * 4;
      float4 va = *(const float4*)(wa + f);
      float4 vb = *(const float4*)(wb + f);
      sBW[wave][r * FEA + f + 0] = va.x * vb.x * scale;
      sBW[wave][r * FEA + f + 1] = va.y * vb.y * scale;
      sBW[wave][r * FEA + f + 2] = va.z * vb.z * scale;
      sBW[wave][r * FEA + f + 3] = va.w * vb.w * scale;
    }
  }
  __syncthreads();
  v8f core[4], gate[4];
  gmlp_branch<256>(&sX[wave][0], &sH[wave][0], WTc0, bc0, WTc1, bc1, lane, core);
  gmlp_branch<256>(&sX[wave][0], &sH[wave][0], WTg0, gb0, WTg1, gb1, lane, gate);
  int m0 = (lane >> 4) * 8, fl = lane & 15;
  #pragma unroll
  for (int nt = 0; nt < 4; ++nt) {
    #pragma unroll
    for (int v = 0; v < 8; ++v) {
      int M = m0 + v, f = nt * 16 + fl;
      float c = core[nt][v]; c = c > 0.f ? c : 0.f;
      float s = 1.f / (1.f + expf(-gate[nt][v]));
      float m = c * s * sBW[wave][M * FEA + f];
      atomicAdd(&new_bond[(size_t)sCtr[wave][M] * FEA + f], m);
    }
  }
}

// ------------------------------------------------------------- angle update
__global__ __launch_bounds__(128) void angle_upd_kernel(
    const float* __restrict__ bond_feas, float* __restrict__ angle_feas,
    const float* __restrict__ atom_feas,
    const int* __restrict__ bg_i, const int* __restrict__ bg_j,
    const int* __restrict__ bg_c,
    const void* WTc, const float* bc, const void* WTg, const float* gb) {
  __shared__ __align__(16) _Float16 sX[MSG_WAVES][16 * 256];
  const int wave = threadIdx.x >> 5;
  const int lane = threadIdx.x & 31;
  const int row0 = (blockIdx.x * MSG_WAVES + wave) * 16;
  {
    int r = lane >> 1, h = lane & 1;
    int row = row0 + r;
    int rc = row < N_ANGLES ? row : 0;
    int bi = bg_i[rc], bj = bg_j[rc], ct = bg_c[rc];
    const float* s0 = bond_feas  + (size_t)bi * FEA;
    const float* s1 = bond_feas  + (size_t)bj * FEA;
    const float* s2 = angle_feas + (size_t)rc * FEA;
    const float* s3 = atom_feas  + (size_t)ct * FEA;
    _Float16* dst = &sX[wave][r * 256];
    #pragma unroll
    for (int j = 0; j < 32; ++j) {
      int f = h * 128 + j * 4;
      const float* sp = (f < 64) ? (s0 + f)
                      : (f < 128) ? (s1 + f - 64)
                      : (f < 192) ? (s2 + f - 128) : (s3 + f - 192);
      cvt_store4(dst, f, *(const float4*)sp);
    }
  }
  __syncthreads();
  v8f ac[4], ag[4];
  #pragma unroll
  for (int nt = 0; nt < 4; ++nt) { ac[nt] = vzero8(); ag[nt] = vzero8(); }
  #pragma unroll
  for (int kc = 0; kc < 8; ++kc) {
    v16h a = lds_afrag(&sX[wave][0], 256, kc * 32, lane);
    #pragma unroll
    for (int nt = 0; nt < 4; ++nt) {
      ac[nt] = wmma32(a, glb_bfrag(WTc, 256, nt, kc * 32, lane), ac[nt]);
      ag[nt] = wmma32(a, glb_bfrag(WTg, 256, nt, kc * 32, lane), ag[nt]);
    }
  }
  int m0 = (lane >> 4) * 8, fl = lane & 15;
  #pragma unroll
  for (int nt = 0; nt < 4; ++nt) {
    #pragma unroll
    for (int v = 0; v < 8; ++v) {
      int M = m0 + v, row = row0 + M;
      if (row < N_ANGLES) {
        int f = nt * 16 + fl;
        float c = ac[nt][v] + bc[f]; c = c > 0.f ? c : 0.f;
        float s = 1.f / (1.f + expf(-(ag[nt][v] + gb[f])));
        size_t idx = (size_t)row * FEA + f;
        angle_feas[idx] = c * s + angle_feas[idx];
      }
    }
  }
}

// ------------------------------------- generic 64x64 linear (+bias/relu/res)
__global__ __launch_bounds__(128) void linear_kernel(
    const float* __restrict__ X, const void* WTv, const float* __restrict__ bias,
    const float* residual, float* __restrict__ out, int n, int do_relu) {
  __shared__ __align__(16) _Float16 sX[MSG_WAVES][16 * FEA];
  const int wave = threadIdx.x >> 5;
  const int lane = threadIdx.x & 31;
  const int row0 = (blockIdx.x * MSG_WAVES + wave) * 16;
  {
    int r = lane >> 1, h = lane & 1;
    int row = row0 + r;
    int rc = row < n ? row : 0;
    const float* sp = X + (size_t)rc * FEA + h * 32;
    _Float16* dst = &sX[wave][r * FEA + h * 32];
    #pragma unroll
    for (int j = 0; j < 8; ++j)
      cvt_store4(dst, j * 4, *(const float4*)(sp + j * 4));
  }
  __syncthreads();
  v8f acc[4];
  #pragma unroll
  for (int nt = 0; nt < 4; ++nt) acc[nt] = vzero8();
  #pragma unroll
  for (int kc = 0; kc < 2; ++kc) {
    v16h a = lds_afrag(&sX[wave][0], FEA, kc * 32, lane);
    #pragma unroll
    for (int nt = 0; nt < 4; ++nt)
      acc[nt] = wmma32(a, glb_bfrag(WTv, FEA, nt, kc * 32, lane), acc[nt]);
  }
  int m0 = (lane >> 4) * 8, fl = lane & 15;
  #pragma unroll
  for (int nt = 0; nt < 4; ++nt) {
    #pragma unroll
    for (int v = 0; v < 8; ++v) {
      int M = m0 + v, row = row0 + M;
      if (row < n) {
        int f = nt * 16 + fl;
        float x = acc[nt][v] + bias[f];
        if (do_relu) x = x > 0.f ? x : 0.f;
        if (residual) x += residual[(size_t)row * FEA + f];
        out[(size_t)row * FEA + f] = x;
      }
    }
  }
}

// ------------------------------------------------------------------ readout
__global__ void readout_kernel(const float* __restrict__ t2,
                               const float* __restrict__ W3,
                               const float* __restrict__ b3,
                               const int* __restrict__ owners,
                               float* __restrict__ e, float* __restrict__ cnt) {
  int a = blockIdx.x * blockDim.x + threadIdx.x;
  if (a >= N_ATOMS) return;
  float acc = b3[0];
  #pragma unroll 8
  for (int f = 0; f < FEA; ++f) acc += t2[(size_t)a * FEA + f] * W3[f];
  atomicAdd(&e[owners[a]], acc);
  atomicAdd(&cnt[owners[a]], 1.0f);
}

__global__ void finalize_kernel(const float* __restrict__ e,
                                const float* __restrict__ cnt,
                                float* __restrict__ out) {
  int g = blockIdx.x * blockDim.x + threadIdx.x;
  if (g < N_GRAPHS) out[g] = e[g] / fmaxf(cnt[g], 1.0f);
}

// =========================================================== host orchestration
extern "C" void kernel_launch(void* const* d_in, const int* in_sizes, int n_in,
                              void* d_out, int out_size, void* d_ws, size_t ws_size,
                              hipStream_t stream) {
  (void)in_sizes; (void)n_in; (void)out_size; (void)ws_size;
  const int*   atomic_numbers = (const int*)d_in[0];
  const float* bond_dist      = (const float*)d_in[1];
  const float* angles         = (const float*)d_in[2];
  const int*   atom_graph     = (const int*)d_in[3];
  const int*   d2u            = (const int*)d_in[4];
  const int*   bg_c           = (const int*)d_in[5];
  const int*   bg_i           = (const int*)d_in[6];
  const int*   bg_j           = (const int*)d_in[7];
  const int*   atom_owners    = (const int*)d_in[8];
  // params: dict insertion order, depth-first
  const float* atom_emb    = (const float*)d_in[10];
  const float* rbf_freq_ag = (const float*)d_in[11];
  const float* rbf_freq_bg = (const float*)d_in[12];
  const float* fourierf    = (const float*)d_in[13];
  const float* bond_emb_W  = (const float*)d_in[14];
  const float* bw_ag_W     = (const float*)d_in[15];
  const float* bw_bg_W     = (const float*)d_in[16];
  const float* angle_emb_W = (const float*)d_in[17];
  auto P = [&](int i) { return (const float*)d_in[i]; };
  const int AC = 18, BC = 58, AL = 88, RO = 100;

  // workspace carve
  char* ws = (char*)d_ws;
  size_t off = 0;
  auto carve = [&](size_t bytes) -> void* {
    void* p = ws + off; off += (bytes + 255) & ~(size_t)255; return p;
  };
  float* atom_feas  = (float*)carve((size_t)N_ATOMS  * FEA * 4);
  float* new_atom   = (float*)carve((size_t)N_ATOMS  * FEA * 4);
  float* bond_feas  = (float*)carve((size_t)N_UBONDS * FEA * 4);
  float* new_bond   = (float*)carve((size_t)N_UBONDS * FEA * 4);
  float* bw_ag      = (float*)carve((size_t)N_UBONDS * FEA * 4);
  float* bw_bg      = (float*)carve((size_t)N_UBONDS * FEA * 4);
  float* angle_feas = (float*)carve((size_t)N_ANGLES * FEA * 4);
  float* t1         = (float*)carve((size_t)N_ATOMS  * FEA * 4);
  float* t2         = (float*)carve((size_t)N_ATOMS  * FEA * 4);
  float* e_g        = (float*)carve(N_GRAPHS * 4);
  float* c_g        = (float*)carve(N_GRAPHS * 4);
  char*  h16        = (char*)carve(1 << 20);   // f16 transposed weights

  // weight conversion table
  WCvtTable tab; tab.n = 0;
  size_t hoff = 0;
  auto addw = [&](const float* src, int din, int dout) -> unsigned {
    unsigned o = (unsigned)hoff;
    tab.src[tab.n] = src; tab.dstOff[tab.n] = o;
    tab.din[tab.n] = din; tab.dout[tab.n] = dout; tab.n++;
    hoff += (size_t)din * dout * 2;
    return o;
  };
  unsigned acW0[4], acW1[4], agW0[4], agW1[4], aoW[4];
  for (int i = 0; i < 4; ++i) {
    int b = AC + 10 * i;
    acW0[i] = addw(P(b + 0), 192, 64);
    acW1[i] = addw(P(b + 2),  64, 64);
    agW0[i] = addw(P(b + 4), 192, 64);
    agW1[i] = addw(P(b + 6),  64, 64);
    aoW [i] = addw(P(b + 8),  64, 64);
  }
  unsigned bcW0[3], bcW1[3], bgW0[3], bgW1[3], boW[3], anC[3], anG[3];
  for (int i = 0; i < 3; ++i) {
    int b = BC + 10 * i;
    bcW0[i] = addw(P(b + 0), 256, 64);
    bcW1[i] = addw(P(b + 2),  64, 64);
    bgW0[i] = addw(P(b + 4), 256, 64);
    bgW1[i] = addw(P(b + 6),  64, 64);
    boW [i] = addw(P(b + 8),  64, 64);
    int a = AL + 4 * i;
    anC[i] = addw(P(a + 0), 256, 64);
    anG[i] = addw(P(a + 2), 256, 64);
  }
  unsigned r0 = addw(P(RO + 0), 64, 64);
  unsigned r1 = addw(P(RO + 2), 64, 64);

  wcvt_kernel<<<tab.n, 256, 0, stream>>>(tab, (void*)h16);

  atom_embed_kernel<<<(N_ATOMS * FEA + 255) / 256, 256, 0, stream>>>(
      atomic_numbers, atom_emb, atom_feas);
  bond_basis_kernel<<<(N_UBONDS * FEA + 255) / 256, 256, 0, stream>>>(
      bond_dist, rbf_freq_ag, rbf_freq_bg, bond_emb_W, bw_ag_W, bw_bg_W,
      bond_feas, bw_ag, bw_bg);
  angle_basis_kernel<<<(N_ANGLES * FEA + 255) / 256, 256, 0, stream>>>(
      angles, fourierf, angle_emb_W, angle_feas);

  const int dblk = ((N_DBONDS + 15) / 16 + MSG_WAVES - 1) / MSG_WAVES;  // 6250
  const int ablk = ((N_ANGLES + 15) / 16 + MSG_WAVES - 1) / MSG_WAVES;  // 6250
  const int lA   = ((N_ATOMS  + 15) / 16 + MSG_WAVES - 1) / MSG_WAVES;  // 313
  const int lB   = ((N_UBONDS + 15) / 16 + MSG_WAVES - 1) / MSG_WAVES;  // 3125

  for (int i = 0; i < 4; ++i) {
    int b = AC + 10 * i;
    hipMemsetAsync(new_atom, 0, (size_t)N_ATOMS * FEA * 4, stream);
    atom_msg_kernel<<<dblk, 128, 0, stream>>>(
        atom_feas, bond_feas, bw_ag, atom_graph, d2u,
        h16 + acW0[i], P(b + 1), h16 + acW1[i], P(b + 3),
        h16 + agW0[i], P(b + 5), h16 + agW1[i], P(b + 7), new_atom);
    linear_kernel<<<lA, 128, 0, stream>>>(
        new_atom, h16 + aoW[i], P(b + 9), atom_feas, atom_feas, N_ATOMS, 0);
    if (i < 3) {
      int bb = BC + 10 * i, ab = AL + 4 * i;
      hipMemsetAsync(new_bond, 0, (size_t)N_UBONDS * FEA * 4, stream);
      bond_msg_kernel<<<ablk, 128, 0, stream>>>(
          bond_feas, angle_feas, atom_feas, bw_bg, bg_i, bg_j, bg_c,
          h16 + bcW0[i], P(bb + 1), h16 + bcW1[i], P(bb + 3),
          h16 + bgW0[i], P(bb + 5), h16 + bgW1[i], P(bb + 7), new_bond);
      linear_kernel<<<lB, 128, 0, stream>>>(
          new_bond, h16 + boW[i], P(bb + 9), bond_feas, bond_feas, N_UBONDS, 0);
      angle_upd_kernel<<<ablk, 128, 0, stream>>>(
          bond_feas, angle_feas, atom_feas, bg_i, bg_j, bg_c,
          h16 + anC[i], P(ab + 1), h16 + anG[i], P(ab + 3));
    }
  }

  linear_kernel<<<lA, 128, 0, stream>>>(
      atom_feas, h16 + r0, P(RO + 1), nullptr, t1, N_ATOMS, 1);
  linear_kernel<<<lA, 128, 0, stream>>>(
      t1, h16 + r1, P(RO + 3), nullptr, t2, N_ATOMS, 1);
  hipMemsetAsync(e_g, 0, N_GRAPHS * 4, stream);
  hipMemsetAsync(c_g, 0, N_GRAPHS * 4, stream);
  readout_kernel<<<(N_ATOMS + 255) / 256, 256, 0, stream>>>(
      t2, P(RO + 4), P(RO + 5), atom_owners, e_g, c_g);
  finalize_kernel<<<1, 256, 0, stream>>>(e_g, c_g, (float*)d_out);
}
